// PositionAwareCausalAttention_40192303956219
// MI455X (gfx1250) — matrix operand order repaired and checked by reading, not compiled, not run on hardware
//
#include <hip/hip_runtime.h>
#include <hip/hip_bf16.h>

// Problem constants (match reference)
#define BB 4
#define TT 2048
#define DD 1024
#define HH 16
#define HD 64

typedef __attribute__((ext_vector_type(16))) __bf16 v16bf;
typedef __attribute__((ext_vector_type(8)))  float  v8f;

union Frag { v16bf v; unsigned u[8]; };

__device__ __forceinline__ unsigned short f2bf(float f) {
  union { float f; unsigned u; } c; c.f = f;
  unsigned r = (c.u + 0x7FFFu + ((c.u >> 16) & 1u)) >> 16;  // RNE
  return (unsigned short)r;
}
__device__ __forceinline__ unsigned pk2(float lo, float hi) {
  return (unsigned)f2bf(lo) | ((unsigned)f2bf(hi) << 16);
}
// K-index within a 16-bit 16x32 A fragment for register v, half = (lane>=16)
__device__ __forceinline__ int kA(int v, int half) {
  return half * 16 + v * 2;
}
__device__ __forceinline__ v8f zero8() {
  v8f z;
#pragma unroll
  for (int i = 0; i < 8; ++i) z[i] = 0.0f;
  return z;
}
// LDS byte offset of a __shared__ object (generic LDS addr low 32 bits are the
// LDS offset per CDNA5 aperture rules).
__device__ __forceinline__ unsigned lds_off(const void* p) {
  return (unsigned)(unsigned long long)p;
}
// CDNA5 async copy global -> LDS, 16B per lane, tracked by ASYNCcnt.
__device__ __forceinline__ void async_g2l_b128(unsigned lds, const void* gaddr) {
  asm volatile("global_load_async_to_lds_b128 %0, %1, off"
               :: "v"(lds), "v"((unsigned long long)gaddr) : "memory");
}
__device__ __forceinline__ void wait_async0() {
  asm volatile("s_wait_asynccnt 0x0" ::: "memory");
}

// ---------------------------------------------------------------------------
// Kernel 1: QKV projection.  C[8192 x 3072] = x[8192 x 1024] @ Wqkv, bf16 WMMA.
// Q,K -> (B,H,T,HD) bf16; V -> (B,H,HD,T) bf16 (pre-transposed for attention's
// async staging + contiguous B-fragment packing).
// ---------------------------------------------------------------------------
__global__ __launch_bounds__(256) void qkv_gemm(
    const float* __restrict__ x, const float* __restrict__ W,
    unsigned short* __restrict__ q_ws, unsigned short* __restrict__ k_ws,
    unsigned short* __restrict__ v_ws) {
  __shared__ __align__(16) unsigned short wsmem[64][32];  // [n][k] bf16
  const int tid = threadIdx.x;
  const int wave = tid >> 5, lane = tid & 31;
  const int half = lane >> 4, col = lane & 15;
  const int m0 = blockIdx.x * 128 + wave * 16;  // this wave's 16 rows
  const int n0 = blockIdx.y * 64;

  v8f acc[4] = {zero8(), zero8(), zero8(), zero8()};

  for (int k0 = 0; k0 < DD; k0 += 32) {
    __syncthreads();
    // Stage W chunk (32 x 64) transposed into LDS as bf16: wsmem[n][k]
    for (int e = tid; e < 2048; e += 256) {
      int n = e & 63, k = e >> 6;
      wsmem[n][k] = f2bf(W[(size_t)(k0 + k) * 3072 + (n0 + n)]);
    }
    if (k0 + 32 < DD)  // hint next chunk into cache (global_prefetch_b8)
      __builtin_prefetch(&W[(size_t)(k0 + 32 + (tid >> 6)) * 3072 + n0 + (tid & 63)], 0, 1);
    __syncthreads();

    // A fragment: x rows (fp32 -> bf16), contiguous K pairs
    Frag a;
    const float* xr = x + (size_t)(m0 + col) * DD + k0;
#pragma unroll
    for (int v = 0; v < 8; ++v) {
      int kk = kA(v, half);
      float2 f = *(const float2*)(xr + kk);
      a.u[v] = pk2(f.x, f.y);
    }
#pragma unroll
    for (int nt = 0; nt < 4; ++nt) {
      Frag bfr;
#pragma unroll
      for (int v = 0; v < 8; ++v)
        bfr.u[v] = *(const unsigned*)&wsmem[nt * 16 + col][half * 16 + 2 * v];
      acc[nt] = __builtin_amdgcn_wmma_f32_16x16x32_bf16(
          false, a.v, false, bfr.v, (short)0, acc[nt], false, false);
    }
  }

  // Epilogue: scatter into Q/K (B,H,T,HD) or V (B,H,HD,T), bf16
  const int which = n0 >> 10;          // 0=q 1=k 2=v (64-col tile stays inside one)
  const int h = (n0 >> 6) & (HH - 1);  // constant within the tile
  const int b = m0 >> 11;              // 128-row tile stays inside one batch
  const size_t bhbase = (size_t)(b * HH + h) * TT * HD;
#pragma unroll
  for (int nt = 0; nt < 4; ++nt) {
#pragma unroll
    for (int r = 0; r < 8; ++r) {
      int row = m0 + r + 8 * half;
      int t = row & (TT - 1);
      int hd = nt * 16 + col;
      unsigned short val = f2bf(acc[nt][r]);
      if (which == 0)      q_ws[bhbase + (size_t)t * HD + hd] = val;
      else if (which == 1) k_ws[bhbase + (size_t)t * HD + hd] = val;
      else                 v_ws[bhbase + (size_t)hd * TT + t] = val;  // transposed
    }
  }
}

// ---------------------------------------------------------------------------
// Kernel 2: Flash attention with position-derived causal mask.
// Block = (b, h, 128 query rows), 8 waves x 16 rows.  K/V blocks of 32 keys
// staged in double-buffered LDS via global_load_async_to_lds_b128 (ASYNCcnt),
// overlapping the next block's fetch with WMMA + online softmax on the
// current one.  QK^T and PV via bf16 WMMA with fp32 accumulation.
// ---------------------------------------------------------------------------
__global__ __launch_bounds__(256) void attn(
    const unsigned short* __restrict__ q_ws, const unsigned short* __restrict__ k_ws,
    const unsigned short* __restrict__ v_ws, const int* __restrict__ pos,
    unsigned short* __restrict__ ao) {
  __shared__ __align__(16) unsigned short ksmem[2][32][64];   // [buf][key][hd]
  __shared__ __align__(16) unsigned short vsmem[2][64][32];   // [buf][hd][key]
  __shared__ __align__(16) unsigned short psmem[8][16][32];   // per-wave P scratch

  const int tid = threadIdx.x, wave = tid >> 5, lane = tid & 31;
  const int half = lane >> 4, col = lane & 15;
  const int h = blockIdx.y, b = blockIdx.z;
  const int q0 = blockIdx.x * 128 + wave * 16;
  const size_t bh = (size_t)(b * HH + h) * TT * HD;   // base for K (T,HD) and V (HD,T)
  const float NINF = -__builtin_inff();

  // Issue one 32-key K/V block as async LDS copies (16B per lane per tensor).
  auto issue_kv = [&](int buf, int jblk) {
    const int kj = tid >> 3, kp = (tid & 7) * 8;     // K: 32 rows x 128B
    async_g2l_b128(lds_off(&ksmem[buf][kj][kp]),
                   k_ws + bh + (size_t)(jblk + kj) * HD + kp);
    const int vr = tid >> 2, vp = (tid & 3) * 8;     // V: 64 rows x 64B
    async_g2l_b128(lds_off(&vsmem[buf][vr][vp]),
                   v_ws + bh + (size_t)vr * TT + jblk + vp);
  };

  // Q fragments (16 x 64 -> two 16x32 A fragments), loaded once
  Frag qa[2];
  {
    const unsigned short* qr = q_ws + bh + (size_t)(q0 + col) * HD;
#pragma unroll
    for (int c = 0; c < 2; ++c)
#pragma unroll
      for (int v = 0; v < 8; ++v)
        qa[c].u[v] = *(const unsigned*)(qr + c * 32 + kA(v, half));
  }
  int posq[8];
#pragma unroll
  for (int r = 0; r < 8; ++r) posq[r] = pos[b * TT + q0 + r + 8 * half];

  float mrow[8], lrow[8];
#pragma unroll
  for (int r = 0; r < 8; ++r) { mrow[r] = NINF; lrow[r] = 0.0f; }
  v8f o[4] = {zero8(), zero8(), zero8(), zero8()};

  issue_kv(0, 0);  // prologue: prime buffer 0

  for (int jb = 0; jb < TT / 32; ++jb) {
    const int cur = jb & 1;
    const int j0 = jb * 32;
    wait_async0();     // this wave's copies for buffer `cur` have landed
    __syncthreads();   // everyone's landed; prior compute on other buf is done
    if (jb + 1 < TT / 32) issue_kv(cur ^ 1, j0 + 32);  // overlap next fetch

    // S = Q @ K^T  (two 16x16 tiles over 32 keys, contraction HD=64 = 2 chunks)
    v8f s0 = zero8(), s1 = zero8();
#pragma unroll
    for (int c = 0; c < 2; ++c) {
      Frag b0, b1;
#pragma unroll
      for (int v = 0; v < 8; ++v) {
        b0.u[v] = *(const unsigned*)&ksmem[cur][col][c * 32 + half * 16 + 2 * v];
        b1.u[v] = *(const unsigned*)&ksmem[cur][col + 16][c * 32 + half * 16 + 2 * v];
      }
      s0 = __builtin_amdgcn_wmma_f32_16x16x32_bf16(false, qa[c].v, false, b0.v,
                                                   (short)0, s0, false, false);
      s1 = __builtin_amdgcn_wmma_f32_16x16x32_bf16(false, qa[c].v, false, b1.v,
                                                   (short)0, s1, false, false);
    }

    const int pk0 = pos[b * TT + j0 + col];
    const int pk1 = pos[b * TT + j0 + 16 + col];
    unsigned short* pw = &psmem[wave][0][0];

#pragma unroll
    for (int r = 0; r < 8; ++r) {
      float a0 = s0[r] * 0.125f;  // 1/sqrt(64)
      float a1 = s1[r] * 0.125f;
      if (posq[r] < pk0) a0 = NINF;
      if (posq[r] < pk1) a1 = NINF;
      float mt = fmaxf(a0, a1);
#pragma unroll
      for (int off = 1; off < 16; off <<= 1) mt = fmaxf(mt, __shfl_xor(mt, off, 16));
      float mnew = fmaxf(mrow[r], mt);
      float alpha = (mnew == NINF) ? 1.0f : __expf(mrow[r] - mnew);
      float p0 = (a0 == NINF) ? 0.0f : __expf(a0 - mnew);
      float p1 = (a1 == NINF) ? 0.0f : __expf(a1 - mnew);
      float rs = p0 + p1;
#pragma unroll
      for (int off = 1; off < 16; off <<= 1) rs += __shfl_xor(rs, off, 16);
      lrow[r] = lrow[r] * alpha + rs;
      mrow[r] = mnew;
      o[0][r] *= alpha; o[1][r] *= alpha; o[2][r] *= alpha; o[3][r] *= alpha;
      int prow = r + 8 * half;
      pw[prow * 32 + col]      = f2bf(p0);
      pw[prow * 32 + col + 16] = f2bf(p1);
    }

    // O += P @ V  (A = P 16x32 from per-wave LDS scratch, B = V tiles from LDS)
    Frag pa;
#pragma unroll
    for (int v = 0; v < 8; ++v)
      pa.u[v] = *(const unsigned*)&pw[col * 32 + kA(v, half)];
#pragma unroll
    for (int nt = 0; nt < 4; ++nt) {
      Frag bv;
#pragma unroll
      for (int v = 0; v < 8; ++v)
        bv.u[v] = *(const unsigned*)&vsmem[cur][nt * 16 + col][half * 16 + 2 * v];
      o[nt] = __builtin_amdgcn_wmma_f32_16x16x32_bf16(false, pa.v, false, bv.v,
                                                      (short)0, o[nt], false, false);
    }
  }

  // Normalize (nan_to_num: fully-masked rows have l==0 -> output 0) and store
  float invr[8];
#pragma unroll
  for (int r = 0; r < 8; ++r) invr[r] = (lrow[r] > 0.0f) ? 1.0f / lrow[r] : 0.0f;
#pragma unroll
  for (int nt = 0; nt < 4; ++nt) {
#pragma unroll
    for (int r = 0; r < 8; ++r) {
      int t = q0 + r + 8 * half;
      ao[(size_t)(b * TT + t) * DD + h * HD + nt * 16 + col] = f2bf(o[nt][r] * invr[r]);
    }
  }
}

// ---------------------------------------------------------------------------
// Kernel 3: out = attn_out[8192 x 1024](bf16) @ Wout[1024 x 1024], f32 result.
// ---------------------------------------------------------------------------
__global__ __launch_bounds__(256) void out_gemm(
    const unsigned short* __restrict__ ao, const float* __restrict__ W,
    float* __restrict__ out) {
  __shared__ __align__(16) unsigned short wsmem[64][32];  // [n][k] bf16
  const int tid = threadIdx.x;
  const int wave = tid >> 5, lane = tid & 31;
  const int half = lane >> 4, col = lane & 15;
  const int m0 = blockIdx.x * 128 + wave * 16;
  const int n0 = blockIdx.y * 64;

  v8f acc[4] = {zero8(), zero8(), zero8(), zero8()};

  for (int k0 = 0; k0 < DD; k0 += 32) {
    __syncthreads();
    for (int e = tid; e < 2048; e += 256) {
      int n = e & 63, k = e >> 6;
      wsmem[n][k] = f2bf(W[(size_t)(k0 + k) * DD + (n0 + n)]);
    }
    if (k0 + 32 < DD)
      __builtin_prefetch(&W[(size_t)(k0 + 32 + (tid >> 6)) * DD + n0 + (tid & 63)], 0, 1);
    __syncthreads();

    Frag a;
    const unsigned short* ar = ao + (size_t)(m0 + col) * DD + k0;
#pragma unroll
    for (int v = 0; v < 8; ++v)
      a.u[v] = *(const unsigned*)(ar + kA(v, half));
#pragma unroll
    for (int nt = 0; nt < 4; ++nt) {
      Frag bfr;
#pragma unroll
      for (int v = 0; v < 8; ++v)
        bfr.u[v] = *(const unsigned*)&wsmem[nt * 16 + col][half * 16 + 2 * v];
      acc[nt] = __builtin_amdgcn_wmma_f32_16x16x32_bf16(
          false, a.v, false, bfr.v, (short)0, acc[nt], false, false);
    }
  }
#pragma unroll
  for (int nt = 0; nt < 4; ++nt) {
#pragma unroll
    for (int r = 0; r < 8; ++r) {
      int row = m0 + r + 8 * half;
      out[(size_t)row * DD + n0 + nt * 16 + col] = acc[nt][r];
    }
  }
}

// ---------------------------------------------------------------------------
// Host-side launch.  Workspace layout (bf16, 8,388,608 elems per tensor):
//   q_ws (B,H,T,HD) | k_ws (B,H,T,HD) | v_ws (B,H,HD,T) | ao (B,T,D)  = 64 MB
// ---------------------------------------------------------------------------
extern "C" void kernel_launch(void* const* d_in, const int* in_sizes, int n_in,
                              void* d_out, int out_size, void* d_ws, size_t ws_size,
                              hipStream_t stream) {
  (void)in_sizes; (void)n_in; (void)out_size; (void)ws_size;
  const float* x    = (const float*)d_in[0];
  const int*   pos  = (const int*)d_in[1];
  const float* Wqkv = (const float*)d_in[2];
  const float* Wout = (const float*)d_in[3];
  float* out = (float*)d_out;

  const size_t per = (size_t)BB * HH * TT * HD;  // 8,388,608 elems
  unsigned short* q_ws = (unsigned short*)d_ws;
  unsigned short* k_ws = q_ws + per;
  unsigned short* v_ws = k_ws + per;
  unsigned short* ao   = v_ws + per;

  dim3 g1(64, 48);      // 8192/128 x 3072/64
  qkv_gemm<<<g1, 256, 0, stream>>>(x, Wqkv, q_ws, k_ws, v_ws);
  dim3 g2(16, HH, BB);  // 2048/128 query tiles x heads x batch
  attn<<<g2, 256, 0, stream>>>(q_ws, k_ws, v_ws, pos, ao);
  dim3 g3(64, 16);      // 8192/128 x 1024/64
  out_gemm<<<g3, 256, 0, stream>>>(ao, Wout, out);
}
